// CARCAttentionProcessor_12610023981795
// MI455X (gfx1250) — matrix operand order repaired and checked by reading, not compiled, not run on hardware
//
#include <hip/hip_runtime.h>
#include <hip/hip_bf16.h>
#include <cmath>
#include <cstdint>

// ---------------------------------------------------------------------------
// CARC attention for MI455X (gfx1250, wave32, WMMA).
// f32->f16 conversions -> 3x WMMA GEMM (QKV, head layouts) ->
// flash-attention with double-buffered ASYNC global->LDS K/V staging
// (GLOBAL_LOAD_ASYNC_TO_LDS_B128 + s_wait_asynccnt), online softmax,
// WMMA S and O -> WMMA GEMM output projection (+bias) in f32.
// ---------------------------------------------------------------------------

typedef _Float16 f16;
typedef __attribute__((ext_vector_type(16))) _Float16 v16h;
typedef __attribute__((ext_vector_type(8)))  _Float16 v8h;
typedef __attribute__((ext_vector_type(8)))  float    v8f;

#define LQ 4096          // query length
#define CH 640           // channels
#define NH 10            // heads
#define HD 64            // head dim
#define LK 8192          // key length after inject (2*LQ)
#define ALPHA_F 0.42f
#define SM_SCALE 0.125f  // 1/sqrt(64)

// ---------------- conversion kernels ----------------

__global__ void cvt_f32_to_f16(const float* __restrict__ src, f16* __restrict__ dst, int n) {
    for (int i = blockIdx.x * blockDim.x + threadIdx.x; i < n; i += gridDim.x * blockDim.x)
        dst[i] = (f16)src[i];
}

// Wt[n][k] = W[k][n]  (640x640)
__global__ void cvt_weight_T(const float* __restrict__ w, f16* __restrict__ wt) {
    int i = blockIdx.x * blockDim.x + threadIdx.x;
    if (i >= CH * CH) return;
    int n = i / CH, k = i % CH;
    wt[i] = (f16)w[k * CH + n];
}

// Fold ALPHA into background KV; K row-major [h][key][d], V transposed [h][d][key].
__global__ void cvt_bg(const float* __restrict__ kbg, const float* __restrict__ vbg,
                       f16* __restrict__ Kh, f16* __restrict__ Vt) {
    int i = blockIdx.x * blockDim.x + threadIdx.x;
    if (i >= NH * LQ * HD) return;
    int h = i / (LQ * HD);
    int r = i % (LQ * HD);
    int q = r / HD, d = r % HD;
    Kh[((size_t)h * LK + LQ + q) * HD + d] = (f16)(ALPHA_F * kbg[i]);
    Vt[((size_t)h * HD + d) * LK + LQ + q] = (f16)(ALPHA_F * vbg[i]);
}

// ---------------- generic WMMA GEMM: Y[M,640] = X[M,640] @ Wt^T ----------------
// BM=128 BN=64 BK=32; 8 waves/block, each wave owns a 16x64 strip.
// mode 0: Q heads Qh[h][q][d] | mode 1: K heads Kh[h][q][d] (first LQ of LK)
// mode 2: V transposed Vt[h][d][q] | mode 3: f32 out + bias.

__global__ __launch_bounds__(256)
void gemm_wmma(const f16* __restrict__ X, const f16* __restrict__ Wt, int mode,
               f16* __restrict__ outQ, f16* __restrict__ outK, f16* __restrict__ outV,
               float* __restrict__ outF, const float* __restrict__ bias) {
    __shared__ f16 sA[128][32];   // 8 KB
    __shared__ f16 sB[64][32];    // 4 KB

    const int tid  = threadIdx.x;
    const int lane = tid & 31;
    const int wave = tid >> 5;
    const int m0   = blockIdx.y * 128;
    const int n0   = blockIdx.x * 64;

    v8f acc[4] = {{}, {}, {}, {}};

    const int ra   = lane & 15;
    const int ab   = (lane < 16) ? 0 : 16;   // A-frag K sub-offset
    const int boff = (lane < 16) ? 0 : 16;   // B-frag K offset

    for (int k0 = 0; k0 < CH; k0 += 32) {
        __syncthreads();
        {   // stage A tile (128x32) and B tile (64x32)
            int row = tid >> 1, seg = tid & 1;
            *(v16h*)&sA[row][seg * 16] =
                *(const v16h*)&X[(size_t)(m0 + row) * CH + k0 + seg * 16];
            if (tid < 128) {
                int n = tid >> 1;
                *(v16h*)&sB[n][seg * 16] =
                    *(const v16h*)&Wt[(size_t)(n0 + n) * CH + k0 + seg * 16];
            }
        }
        __syncthreads();

        union { v16h v; v8h h[2]; } ua;  // A frag: contiguous k = 16*(lane/16) + s
        ua.h[0] = *(const v8h*)&sA[wave * 16 + ra][ab];
        ua.h[1] = *(const v8h*)&sA[wave * 16 + ra][ab + 8];
        #pragma unroll
        for (int j = 0; j < 4; ++j) {
            v16h bv = *(const v16h*)&sB[j * 16 + ra][boff];
            acc[j] = __builtin_amdgcn_wmma_f32_16x16x32_f16(
                false, ua.v, false, bv, (short)0, acc[j], false, false);
        }
    }

    const int colb = lane & 15;
    const int rofs = (lane < 16) ? 0 : 8;
    #pragma unroll
    for (int j = 0; j < 4; ++j) {
        #pragma unroll
        for (int r = 0; r < 8; ++r) {
            int m = m0 + wave * 16 + rofs + r;
            int n = n0 + j * 16 + colb;
            float v = acc[j][r];
            if (mode == 0) {
                int h = n >> 6, d = n & 63;
                outQ[((size_t)h * LQ + m) * HD + d] = (f16)v;
            } else if (mode == 1) {
                int h = n >> 6, d = n & 63;
                outK[((size_t)h * LK + m) * HD + d] = (f16)v;
            } else if (mode == 2) {
                int h = n >> 6, d = n & 63;
                outV[((size_t)h * HD + d) * LK + m] = (f16)v;
            } else {
                outF[(size_t)m * CH + n] = v + bias[n];
            }
        }
    }
}

// ---------------- flash attention ----------------
// Block: 1 head x 128 queries (8 waves x 16 rows). K/V chunks (32 keys) are
// staged into double-buffered LDS with CDNA5 async global->LDS copies
// (ASYNCcnt); chunk i+1 is in flight while WMMAs consume chunk i.

__global__ __launch_bounds__(256)
void attn_wmma(const f16* __restrict__ Qh, const f16* __restrict__ Kh,
               const f16* __restrict__ Vt, f16* __restrict__ Ctx) {
    __shared__ f16 sK[2][32][64];     // 2 x 4 KB, [key][d]
    __shared__ f16 sV[2][64][32];     // 2 x 4 KB, [d][key]
    __shared__ f16 sP[8][16][32];     // 8 KB, per-wave P reshape patch

    const int tid  = threadIdx.x;
    const int lane = tid & 31;
    const int wave = tid >> 5;
    const int head = blockIdx.y;
    const int q0   = blockIdx.x * 128 + wave * 16;

    const int ra   = lane & 15;
    const int ab   = (lane < 16) ? 0 : 16;
    const int boff = (lane < 16) ? 0 : 16;

    const f16* Kbase = &Kh[(size_t)head * LK * HD];
    const f16* Vbase = &Vt[(size_t)head * HD * LK];
    const int vr = tid >> 2;            // V staging row in [0,64)
    const int vc = (tid & 3) * 8;       // V staging col chunk

    // LDS byte offsets for this thread's 16B async slice (flat offset == tid*16
    // for both K[32][64] and V[64][32] staging maps).
    const unsigned ldsK0 = (unsigned)(uintptr_t)(&sK[0][0][0]) + (unsigned)(tid * 16);
    const unsigned ldsK1 = (unsigned)(uintptr_t)(&sK[1][0][0]) + (unsigned)(tid * 16);
    const unsigned ldsV0 = (unsigned)(uintptr_t)(&sV[0][0][0]) + (unsigned)(tid * 16);
    const unsigned ldsV1 = (unsigned)(uintptr_t)(&sV[1][0][0]) + (unsigned)(tid * 16);

    auto issue_chunk = [&](int buf, int kb) {
        const f16* gk = Kbase + (size_t)kb * HD + tid * 8;           // contiguous chunk
        const f16* gv = Vbase + (size_t)vr * LK + kb + vc;           // strided rows
        unsigned lk = buf ? ldsK1 : ldsK0;
        unsigned lv = buf ? ldsV1 : ldsV0;
        asm volatile("global_load_async_to_lds_b128 %0, %1, off"
                     :: "v"(lk), "v"(gk) : "memory");
        asm volatile("global_load_async_to_lds_b128 %0, %1, off"
                     :: "v"(lv), "v"(gv) : "memory");
    };

    // Q fragments for this wave's 16 rows, scaled by 1/sqrt(D)
    union { v16h v; v8h h[2]; } qa[2];
    {
        const f16* qrow = &Qh[((size_t)head * LQ + q0 + ra) * HD];
        #pragma unroll
        for (int f = 0; f < 2; ++f) {
            qa[f].h[0] = *(const v8h*)(qrow + f * 32 + ab);
            qa[f].h[1] = *(const v8h*)(qrow + f * 32 + ab + 8);
            #pragma unroll
            for (int i = 0; i < 16; ++i)
                qa[f].v[i] = (f16)((float)qa[f].v[i] * SM_SCALE);
        }
    }

    v8f o[4] = {{}, {}, {}, {}};
    float mrow[8], lrow[8];
    #pragma unroll
    for (int r = 0; r < 8; ++r) { mrow[r] = -1e30f; lrow[r] = 0.0f; }

    issue_chunk(0, 0);   // preload first chunk

    for (int kb = 0; kb < LK; kb += 32) {
        const int cur = (kb >> 5) & 1;
        // Current chunk's async copies done (per-wave), then barrier: previous
        // buffer reads finished (WMMA forced s_wait_dscnt pre-barrier), so the
        // next issue below can safely overwrite it.
        asm volatile("s_wait_asynccnt 0x0" ::: "memory");
        __syncthreads();
        if (kb + 32 < LK) issue_chunk(cur ^ 1, kb + 32);

        // S tiles: c0 = keys kb..kb+15, c1 = keys kb+16..kb+31 (K-dim = 64)
        v8f c0 = {}, c1 = {};
        {
            v16h b00 = *(const v16h*)&sK[cur][ra][boff];
            v16h b01 = *(const v16h*)&sK[cur][ra][32 + boff];
            v16h b10 = *(const v16h*)&sK[cur][16 + ra][boff];
            v16h b11 = *(const v16h*)&sK[cur][16 + ra][32 + boff];
            c0 = __builtin_amdgcn_wmma_f32_16x16x32_f16(false, qa[0].v, false, b00, (short)0, c0, false, false);
            c0 = __builtin_amdgcn_wmma_f32_16x16x32_f16(false, qa[1].v, false, b01, (short)0, c0, false, false);
            c1 = __builtin_amdgcn_wmma_f32_16x16x32_f16(false, qa[0].v, false, b10, (short)0, c1, false, false);
            c1 = __builtin_amdgcn_wmma_f32_16x16x32_f16(false, qa[1].v, false, b11, (short)0, c1, false, false);
        }

        // Online softmax. C-layout: row = r + 8*(lane>=16), col = lane%16;
        // row reductions are 16-lane xor butterflies.
        float p0[8], p1[8], al[8];
        #pragma unroll
        for (int r = 0; r < 8; ++r) {
            float t = fmaxf(c0[r], c1[r]);
            #pragma unroll
            for (int s = 1; s < 16; s <<= 1)
                t = fmaxf(t, __shfl_xor(t, s, 32));
            float mn = fmaxf(mrow[r], t);
            al[r] = __expf(mrow[r] - mn);
            mrow[r] = mn;
            p0[r] = __expf(c0[r] - mn);
            p1[r] = __expf(c1[r] - mn);
            float rs = p0[r] + p1[r];
            #pragma unroll
            for (int s = 1; s < 16; s <<= 1)
                rs += __shfl_xor(rs, s, 32);
            lrow[r] = lrow[r] * al[r] + rs;
        }
        #pragma unroll
        for (int j = 0; j < 4; ++j)
            #pragma unroll
            for (int r = 0; r < 8; ++r)
                o[j][r] *= al[r];

        // Reshape P (C-layout) -> A-fragment via per-wave LDS patch.
        {
            int prow = (lane < 16) ? 0 : 8;
            #pragma unroll
            for (int r = 0; r < 8; ++r) {
                sP[wave][prow + r][ra]      = (f16)p0[r];
                sP[wave][prow + r][ra + 16] = (f16)p1[r];
            }
        }
        union { v16h v; v8h h[2]; } pa;
        pa.h[0] = *(const v8h*)&sP[wave][ra][ab];
        pa.h[1] = *(const v8h*)&sP[wave][ra][ab + 8];

        // O += P @ V (4 d-tiles, K-dim = 32 keys)
        #pragma unroll
        for (int j = 0; j < 4; ++j) {
            v16h bv = *(const v16h*)&sV[cur][j * 16 + ra][boff];
            o[j] = __builtin_amdgcn_wmma_f32_16x16x32_f16(
                false, pa.v, false, bv, (short)0, o[j], false, false);
        }
    }

    // Normalize and write ctx in [L, C] layout (col = head*64 + d)
    #pragma unroll
    for (int j = 0; j < 4; ++j) {
        #pragma unroll
        for (int r = 0; r < 8; ++r) {
            int m   = q0 + ((lane < 16) ? 0 : 8) + r;
            int col = head * HD + j * 16 + ra;
            Ctx[(size_t)m * CH + col] = (f16)(o[j][r] / lrow[r]);
        }
    }
}

// ---------------- launch ----------------

extern "C" void kernel_launch(void* const* d_in, const int* in_sizes, int n_in,
                              void* d_out, int out_size, void* d_ws, size_t ws_size,
                              hipStream_t stream) {
    (void)in_sizes; (void)n_in; (void)out_size; (void)ws_size;

    const float* hs  = (const float*)d_in[0];
    const float* kbg = (const float*)d_in[1];
    const float* vbg = (const float*)d_in[2];
    const float* Wq  = (const float*)d_in[3];
    const float* Wk  = (const float*)d_in[4];
    const float* Wv  = (const float*)d_in[5];
    const float* Wo  = (const float*)d_in[6];
    const float* bo  = (const float*)d_in[7];
    float* out = (float*)d_out;

    // Workspace carve-up (f16, all 32B-aligned). Total ~40 MB.
    char* w = (char*)d_ws;
    f16* Xh  = (f16*)w; w += (size_t)LQ * CH * 2;        // hidden f16
    f16* WqT = (f16*)w; w += (size_t)CH * CH * 2;
    f16* WkT = (f16*)w; w += (size_t)CH * CH * 2;
    f16* WvT = (f16*)w; w += (size_t)CH * CH * 2;
    f16* WoT = (f16*)w; w += (size_t)CH * CH * 2;
    f16* Qh  = (f16*)w; w += (size_t)NH * LQ * HD * 2;   // [h][q][d]
    f16* Kh  = (f16*)w; w += (size_t)NH * LK * HD * 2;   // [h][key][d]
    f16* Vt  = (f16*)w; w += (size_t)NH * HD * LK * 2;   // [h][d][key]
    f16* Ctx = (f16*)w; w += (size_t)LQ * CH * 2;

    // 1) conversions
    cvt_f32_to_f16<<<1024, 256, 0, stream>>>(hs, Xh, LQ * CH);
    cvt_weight_T<<<(CH * CH + 255) / 256, 256, 0, stream>>>(Wq, WqT);
    cvt_weight_T<<<(CH * CH + 255) / 256, 256, 0, stream>>>(Wk, WkT);
    cvt_weight_T<<<(CH * CH + 255) / 256, 256, 0, stream>>>(Wv, WvT);
    cvt_weight_T<<<(CH * CH + 255) / 256, 256, 0, stream>>>(Wo, WoT);
    cvt_bg<<<(NH * LQ * HD + 255) / 256, 256, 0, stream>>>(kbg, vbg, Kh, Vt);

    // 2) QKV projections (WMMA GEMM)
    dim3 gp(CH / 64, LQ / 128);
    gemm_wmma<<<gp, 256, 0, stream>>>(Xh, WqT, 0, Qh, Kh, Vt, out, bo);
    gemm_wmma<<<gp, 256, 0, stream>>>(Xh, WkT, 1, Qh, Kh, Vt, out, bo);
    gemm_wmma<<<gp, 256, 0, stream>>>(Xh, WvT, 2, Qh, Kh, Vt, out, bo);

    // 3) flash attention
    dim3 ga(LQ / 128, NH);
    attn_wmma<<<ga, 256, 0, stream>>>(Qh, Kh, Vt, Ctx);

    // 4) output projection + bias (f32 out)
    gemm_wmma<<<gp, 256, 0, stream>>>(Ctx, WoT, 3, Qh, Kh, Vt, out, bo);
}
